// Net_2327872274899
// MI455X (gfx1250) — compile-verified, matrix operand order checked by reading, not run on hardware
//
#include <hip/hip_runtime.h>
#include <hip/hip_bf16.h>
#include <math.h>

// ---------------------------------------------------------------------------
// Problem constants (match reference)
// ---------------------------------------------------------------------------
#define NNODE 6144
#define NGRAPH 32
#define NPG 192
#define FIN 32
#define FE 8
#define HDIM 64
#define HM 128
#define KNN 8
#define NEDGE 49152
#define NN2 (NPG*NPG)            // 36864
#define R2 (NGRAPH*NN2)          // 1179648 rows of model2 head
#define TILES2 (R2/16)           // 73728
#define BN_EPS 1e-5f

typedef __attribute__((ext_vector_type(16))) _Float16 v16h;
typedef __attribute__((ext_vector_type(8)))  float    v8f;

#define WMMA_F16(A, B, C) \
  __builtin_amdgcn_wmma_f32_16x16x32_f16(false, (A), false, (B), (short)0, (C), false, false)

// K offset within a 32-wide chunk of fragment element-pair t, lane-half hi.
// Branchless: t<4 -> 2t+8hi ; t>=4 -> 16+2(t-4)+8hi == 2t+8+8hi.
__device__ __forceinline__ int frag_k(int t, int hi) {
  return 2 * t + 8 * hi + 8 * (t >> 2);
}

// Fragment-major f16 array: frag index c2, lane -> 16 contiguous halves (32B)
__device__ __forceinline__ v16h load_frag(const _Float16* wf, int c2, int lane) {
  return *(const v16h*)(wf + (((c2 << 5) + lane) << 4));
}
__device__ __forceinline__ v16h load_wf(const _Float16* wf, int nt, int kc, int KC, int lane) {
  return load_frag(wf, nt * KC + kc, lane);
}

// Scatter one D-element (row m, col) into A-fragment-major f16 storage (branchless)
__device__ __forceinline__ void store_frag_elem(_Float16* f, int m, int col, float v) {
  int kc  = col >> 5;
  int kk  = col & 31;
  int hi2 = (kk >> 3) & 1;
  int e   = kk - 8 * hi2 - 8 * (kk >> 4);
  int lane2 = m + 16 * hi2;
  f[(((kc << 5) + lane2) << 4) + e] = (_Float16)v;
}

// Order-preserving float<->uint mapping for atomicMax-based segment_max
__device__ __forceinline__ unsigned f32_ord(float f) {
  unsigned u = __float_as_uint(f);
  return (u & 0x80000000u) ? ~u : (u | 0x80000000u);
}
__device__ __forceinline__ float ord_f32(unsigned m) {
  unsigned u = (m & 0x80000000u) ? (m & 0x7FFFFFFFu) : ~m;
  return __uint_as_float(u);
}

__device__ __forceinline__ void top8_insert(float (&dq)[8], int (&iq)[8], float v, int j) {
  if (v < dq[7]) {
    dq[7] = v; iq[7] = j;
#pragma unroll
    for (int t = 7; t > 0; --t) {
      if (dq[t] < dq[t - 1]) {
        float td = dq[t]; dq[t] = dq[t - 1]; dq[t - 1] = td;
        int   ti = iq[t]; iq[t] = iq[t - 1]; iq[t - 1] = ti;
      }
    }
  }
}

// ---------------------------------------------------------------------------
// Init / swizzle / stats kernels
// ---------------------------------------------------------------------------
__global__ void init_kernel(float* x1acc, unsigned* x2bits, float* msum, float* msumsq) {
  int i = blockIdx.x * blockDim.x + threadIdx.x;
  if (i < NNODE * HDIM) {
    x1acc[i]  = 0.0f;
    x2bits[i] = 0x007FFFFFu;   // f32_ord(-inf)
  }
  if (i < HM) { msum[i] = 0.0f; msumsq[i] = 0.0f; }
}

// W[K][ncols] (rows >= kvalid are zero) -> fragment-major f16
__global__ void swizzle_w_kernel(const float* __restrict__ w, _Float16* __restrict__ wf,
                                 int K, int kvalid, int ncols) {
  int idx = blockIdx.x * blockDim.x + threadIdx.x;
  if (idx >= K * ncols) return;
  int e    = idx & 15;
  int lane = (idx >> 4) & 31;
  int c2   = idx >> 9;
  int KC = K >> 5;
  int nt = c2 / KC, kc = c2 - nt * KC;
  int hi = lane >> 4, lm = lane & 15;
  int n = nt * 16 + lm;
  int t = e >> 1;
  int k = kc * 32 + frag_k(t, hi) + (e & 1);
  wf[idx] = (k < kvalid) ? (_Float16)w[k * ncols + n] : (_Float16)0.0f;
}

// x1n[N][64] -> fragment-major f16 (serves both A- and B-fragments of the KNN GEMM)
__global__ void swizzle_x1_kernel(const float* __restrict__ x1n, _Float16* __restrict__ xf) {
  int idx = blockIdx.x * blockDim.x + threadIdx.x;   // (jb*2 + kc) * 32 + lane
  if (idx >= (NNODE / 16) * 2 * 32) return;
  int lane = idx & 31;
  int c2   = idx >> 5;
  int jb = c2 >> 1, kc = c2 & 1;
  int hi = lane >> 4, lm = lane & 15;
  const float* row = x1n + (jb * 16 + lm) * HDIM + kc * 32;
  _Float16* out = xf + ((size_t)idx << 4);
#pragma unroll
  for (int e = 0; e < 16; ++e) {
    int k = frag_k(e >> 1, hi) + (e & 1);
    out[e] = (_Float16)row[k];
  }
}

// One block per column: biased batch stats (sum, sumsq)
__global__ void colstats_kernel(const float* x, int nrows, int ncols, float* sum, float* sumsq) {
  int c = blockIdx.x;
  float s = 0.0f, q = 0.0f;
  for (int r = threadIdx.x; r < nrows; r += blockDim.x) {
    float v = x[r * ncols + c];
    s += v; q += v * v;
  }
  __shared__ float ss[256], qq[256];
  ss[threadIdx.x] = s; qq[threadIdx.x] = q;
  __syncthreads();
  for (int off = blockDim.x >> 1; off > 0; off >>= 1) {
    if (threadIdx.x < off) { ss[threadIdx.x] += ss[threadIdx.x + off]; qq[threadIdx.x] += qq[threadIdx.x + off]; }
    __syncthreads();
  }
  if (threadIdx.x == 0) { sum[c] = ss[0]; sumsq[c] = qq[0]; }
}

__global__ void bn_finalize_kernel(const float* sum, const float* sumsq,
                                   const float* g, const float* b,
                                   float inv_n, int ncols, float* scale, float* shift) {
  int c = blockIdx.x * blockDim.x + threadIdx.x;
  if (c >= ncols) return;
  float mean = sum[c] * inv_n;
  float var  = sumsq[c] * inv_n - mean * mean;
  float s    = g[c] * rsqrtf(var + BN_EPS);
  scale[c] = s;
  shift[c] = b[c] - mean * s;
}

// ---------------------------------------------------------------------------
// Layer 1: EdgeFeatsConv message MLP (16 edges / wave, WMMA, K padded 72->96)
// ---------------------------------------------------------------------------
// A-fragment for K-chunk kc. All dispatch on the literal kc (caller unrolls);
// the only runtime select is the hi==0 mask in the edge-attr chunk.
__device__ __forceinline__ v16h ef_a_frag(const float* __restrict__ xd_row,
                                          const float* __restrict__ xs_row,
                                          const float* __restrict__ ea_row,
                                          int kc, int hi) {
  v16h a;
  if (kc < 2) {
    const float* row = (kc == 0) ? xd_row : xs_row;
#pragma unroll
    for (int e = 0; e < 16; ++e) {
      int k = frag_k(e >> 1, hi) + (e & 1);
      a[e] = (_Float16)row[k];
    }
  } else {
    // cols 64..95: for hi==0 the nonzero elements are exactly e<8 with k==e.
#pragma unroll
    for (int e = 0; e < 16; ++e) {
      float v = ea_row[e & 7];                       // always-safe load, e literal
      a[e] = ((e < 8) && (hi == 0)) ? (_Float16)v : (_Float16)0.0f;
    }
  }
  return a;
}

__global__ void edgefeat_kernel(const float* __restrict__ x, const int* __restrict__ ei,
                                const float* __restrict__ ea,
                                const _Float16* __restrict__ w1f, const float* __restrict__ b1,
                                const _Float16* __restrict__ w2f, const float* __restrict__ b2,
                                float* __restrict__ x1acc) {
  __shared__ __align__(32) _Float16 AhF[2 * 32 * 16];   // hidden, fragment-major
  __shared__ int dsts[16];
  const int lane = threadIdx.x, hi = lane >> 4, lm = lane & 15;
  const int e = blockIdx.x * 16 + lm;
  const int s = ei[e];
  const int d = ei[NEDGE + e];
  if (hi == 0) dsts[lm] = d;
  const float* xd_row = x + d * FIN;
  const float* xs_row = x + s * FIN;
  const float* ea_row = ea + e * FE;
  // hidden = relu(A @ w1 + b1), scattered into fragment-major LDS
#pragma unroll
  for (int nt = 0; nt < 4; ++nt) {
    v8f acc = {};
#pragma unroll
    for (int kc = 0; kc < 3; ++kc) {
      v16h a = ef_a_frag(xd_row, xs_row, ea_row, kc, hi);
      v16h b = load_wf(w1f, nt, kc, 3, lane);
      acc = WMMA_F16(a, b, acc);
    }
    int col = nt * 16 + lm;
    float bias = b1[col];
#pragma unroll
    for (int rr = 0; rr < 8; ++rr)
      store_frag_elem(AhF, rr + 8 * hi, col, fmaxf(acc[rr] + bias, 0.0f));
  }
  __syncthreads();
  // msg = hidden @ w2 + b2, scatter-add by dst
#pragma unroll
  for (int nt = 0; nt < 4; ++nt) {
    v8f acc = {};
#pragma unroll
    for (int kc = 0; kc < 2; ++kc) {
      v16h a = load_frag(AhF, kc, lane);
      v16h b = load_wf(w2f, nt, kc, 2, lane);
      acc = WMMA_F16(a, b, acc);
    }
    int col = nt * 16 + lm;
    float bias = b2[col];
#pragma unroll
    for (int rr = 0; rr < 8; ++rr) {
      int m = rr + 8 * hi;
      atomicAdd(&x1acc[dsts[m] * HDIM + col], acc[rr] + bias);
    }
  }
}

// x1raw = x1acc + x @ ef_wroot   (16 nodes / wave, K=32)
__global__ void root_kernel(const float* __restrict__ x, const _Float16* __restrict__ wrootf,
                            const float* __restrict__ x1acc, float* __restrict__ x1raw) {
  const int lane = threadIdx.x, hi = lane >> 4, lm = lane & 15;
  const int m0 = blockIdx.x * 16;
  // A fragment direct from global x (row m0+lm)
  v16h a;
  {
    const float* row = x + (m0 + lm) * FIN;
#pragma unroll
    for (int e = 0; e < 16; ++e) {
      int k = frag_k(e >> 1, hi) + (e & 1);
      a[e] = (_Float16)row[k];
    }
  }
#pragma unroll
  for (int nt = 0; nt < 4; ++nt) {
    v8f acc = {};
    v16h b = load_wf(wrootf, nt, 0, 1, lane);
    acc = WMMA_F16(a, b, acc);
    int col = nt * 16 + lm;
#pragma unroll
    for (int rr = 0; rr < 8; ++rr) {
      int m = rr + 8 * hi;
      int idx = (m0 + m) * HDIM + col;
      x1raw[idx] = acc[rr] + x1acc[idx];
    }
  }
}

// Apply BN1; emit f32 x1n and row sq-norms (consistent with f16 rounding)
__global__ void bn_apply_x1_kernel(const float* __restrict__ x1raw,
                                   const float* __restrict__ scale, const float* __restrict__ shift,
                                   float* __restrict__ x1n, float* __restrict__ sqn) {
  int n = blockIdx.x * blockDim.x + threadIdx.x;
  if (n >= NNODE) return;
  float sq = 0.0f;
  for (int c = 0; c < HDIM; ++c) {
    float v = x1raw[n * HDIM + c] * scale[c] + shift[c];
    x1n[n * HDIM + c] = v;
    float fh = (float)(_Float16)v;
    sq += fh * fh;
  }
  sqn[n] = sq;
}

// ---------------------------------------------------------------------------
// KNN: one wave per 16-column (node-i) strip; fragment-major x1 serves A and B
// ---------------------------------------------------------------------------
__global__ void knn_kernel(const _Float16* __restrict__ xf, const float* __restrict__ sqn,
                           int* __restrict__ nbr) {
  const int lane = threadIdx.x, hi = lane >> 4, lm = lane & 15;
  const int ib = blockIdx.x;
  const int icol = ib * 16 + lm;
  v16h B0 = load_frag(xf, ib * 2 + 0, lane);
  v16h B1 = load_frag(xf, ib * 2 + 1, lane);
  const float sqi = sqn[icol];
  float dq[8]; int iq[8];
#pragma unroll
  for (int t = 0; t < 8; ++t) { dq[t] = INFINITY; iq[t] = 0; }

  for (int jb = 0; jb < NNODE / 16; ++jb) {
    v8f acc = {};
    acc = WMMA_F16(load_frag(xf, jb * 2 + 0, lane), B0, acc);
    acc = WMMA_F16(load_frag(xf, jb * 2 + 1, lane), B1, acc);
#pragma unroll
    for (int rr = 0; rr < 8; ++rr) {
      int j = jb * 16 + rr + 8 * hi;
      float d2 = sqn[j] + sqi - 2.0f * acc[rr];
      if (j != icol) top8_insert(dq, iq, d2, j);
    }
  }
  // merge lane pair (rows split across L and L+16)
  float pd[8]; int pi[8];
#pragma unroll
  for (int t = 0; t < 8; ++t) {
    pd[t] = __shfl_xor(dq[t], 16);
    pi[t] = __shfl_xor(iq[t], 16);
  }
#pragma unroll
  for (int t = 0; t < 8; ++t) top8_insert(dq, iq, pd[t], pi[t]);
  if (lane < 16) {
#pragma unroll
    for (int t = 0; t < 8; ++t) nbr[icol * KNN + t] = iq[t];
  }
}

// ---------------------------------------------------------------------------
// Layer 2: EdgeConv MLP over 2*N*K directed contributions (16 / wave), max-aggr
// ---------------------------------------------------------------------------
// A-fragment: kc<2 -> x1[d] cols; kc>=2 -> (x1[s]-x1[d]) cols. Dispatch on the
// literal kc only (k index never compared at runtime).
__device__ __forceinline__ v16h ec_a_frag(const float* __restrict__ xd, const float* __restrict__ xs,
                                          int kc, int hi) {
  v16h a;
  const int koff = (kc & 1) * 32;
#pragma unroll
  for (int e = 0; e < 16; ++e) {
    int k = koff + frag_k(e >> 1, hi) + (e & 1);
    float v = (kc < 2) ? xd[k] : (xs[k] - xd[k]);
    a[e] = (_Float16)v;
  }
  return a;
}

__global__ void edgeconv_kernel(const float* __restrict__ x1n, const int* __restrict__ nbr,
                                const _Float16* __restrict__ w1f, const float* __restrict__ b1,
                                const _Float16* __restrict__ w2f, const float* __restrict__ b2,
                                unsigned* __restrict__ x2bits) {
  __shared__ __align__(32) _Float16 AhF[2 * 32 * 16];
  __shared__ int dsts[16];
  const int lane = threadIdx.x, hi = lane >> 4, lm = lane & 15;
  const int tr = blockIdx.x * 16 + lm;
  const int e = tr >> 1;
  const int i = e >> 3;
  const int q = e & 7;
  const int j = nbr[i * KNN + q];
  const int d = (tr & 1) ? j : i;
  const int s = (tr & 1) ? i : j;
  if (hi == 0) dsts[lm] = d;
  const float* xd = x1n + d * HDIM;
  const float* xs = x1n + s * HDIM;
#pragma unroll
  for (int nt = 0; nt < 4; ++nt) {
    v8f acc = {};
#pragma unroll
    for (int kc = 0; kc < 4; ++kc) {
      v16h a = ec_a_frag(xd, xs, kc, hi);
      v16h b = load_wf(w1f, nt, kc, 4, lane);
      acc = WMMA_F16(a, b, acc);
    }
    int col = nt * 16 + lm;
    float bias = b1[col];
#pragma unroll
    for (int rr = 0; rr < 8; ++rr)
      store_frag_elem(AhF, rr + 8 * hi, col, fmaxf(acc[rr] + bias, 0.0f));
  }
  __syncthreads();
#pragma unroll
  for (int nt = 0; nt < 4; ++nt) {
    v8f acc = {};
#pragma unroll
    for (int kc = 0; kc < 2; ++kc) {
      v16h a = load_frag(AhF, kc, lane);
      v16h b = load_wf(w2f, nt, kc, 2, lane);
      acc = WMMA_F16(a, b, acc);
    }
    int col = nt * 16 + lm;
    float bias = b2[col];
#pragma unroll
    for (int rr = 0; rr < 8; ++rr) {
      int m = rr + 8 * hi;
      atomicMax(&x2bits[dsts[m] * HDIM + col], f32_ord(acc[rr] + bias));
    }
  }
}

__global__ void x2_finalize_kernel(const unsigned* __restrict__ bits, float* __restrict__ x2) {
  int idx = blockIdx.x * blockDim.x + threadIdx.x;
  if (idx < NNODE * HDIM) x2[idx] = ord_f32(bits[idx]);
}

__global__ void bn_apply_x2_kernel(const float* __restrict__ x2,
                                   const float* __restrict__ scale, const float* __restrict__ shift,
                                   float* __restrict__ x2n) {
  int idx = blockIdx.x * blockDim.x + threadIdx.x;
  if (idx >= NNODE * HDIM) return;
  int c = idx & (HDIM - 1);
  x2n[idx] = x2[idx] * scale[c] + shift[c];
}

// ---------------------------------------------------------------------------
// model2 head, fused over |g_i - g_z| rows (row r: g, z = inner-outer, i inner)
// ---------------------------------------------------------------------------
__device__ __forceinline__ void m2_row_ptrs(const float* __restrict__ x2n, int r,
                                            const float*& xi, const float*& xz) {
  int g = r / NN2;
  int rem = r - g * NN2;
  int z = rem / NPG;
  int i = rem - z * NPG;
  xi = x2n + (g * NPG + i) * HDIM;
  xz = x2n + (g * NPG + z) * HDIM;
}

// |xi - xz| -> f16. An empty inline-asm VGPR barrier separates the subtract
// from the abs+convert: the gfx1250 backend crashes when the SLP-vectorized
// v2f32 subtract is folded into an fabs sign-mask feeding the f16 pack
// (malformed v2f32<->i32 bitcast node). The barrier makes the diffs opaque,
// so fabs/cvt lower to plain scalar v_and/v_cvt which are always selectable.
__device__ __forceinline__ v16h m2_a_frag(const float* __restrict__ xi, const float* __restrict__ xz,
                                          int kc, int hi) {
  v16h a;
#pragma unroll
  for (int t = 0; t < 8; ++t) {
    int k = kc * 32 + frag_k(t, hi);
    float d0 = xi[k] - xz[k];
    float d1 = xi[k + 1] - xz[k + 1];
    asm("" : "+v"(d0), "+v"(d1));      // block fabs-mask DAG fold (compiler bug workaround)
    a[2 * t]     = (_Float16)fabsf(d0);
    a[2 * t + 1] = (_Float16)fabsf(d1);
  }
  return a;
}

// Pass 1: column stats of y = diff @ W1 + b1  (no LDS, register accumulation)
__global__ void m2_pass1_kernel(const float* __restrict__ x2n,
                                const _Float16* __restrict__ w1f, const float* __restrict__ b1,
                                float* __restrict__ msum, float* __restrict__ msumsq) {
  const int lane = threadIdx.x, hi = lane >> 4, lm = lane & 15;
  float accS[8], accQ[8];
#pragma unroll
  for (int t = 0; t < 8; ++t) { accS[t] = 0.0f; accQ[t] = 0.0f; }

  for (int tile = blockIdx.x; tile < TILES2; tile += gridDim.x) {
    const float *xi, *xz;
    m2_row_ptrs(x2n, tile * 16 + lm, xi, xz);
    v16h a0 = m2_a_frag(xi, xz, 0, hi);
    v16h a1 = m2_a_frag(xi, xz, 1, hi);
#pragma unroll
    for (int nt = 0; nt < 8; ++nt) {
      v8f acc = {};
      acc = WMMA_F16(a0, load_wf(w1f, nt, 0, 2, lane), acc);
      acc = WMMA_F16(a1, load_wf(w1f, nt, 1, 2, lane), acc);
      float bias = b1[nt * 16 + lm];
#pragma unroll
      for (int rr = 0; rr < 8; ++rr) {
        float y = acc[rr] + bias;
        accS[nt] += y;
        accQ[nt] += y * y;
      }
    }
  }
#pragma unroll
  for (int nt = 0; nt < 8; ++nt) {
    float s = accS[nt] + __shfl_xor(accS[nt], 16);
    float q = accQ[nt] + __shfl_xor(accQ[nt], 16);
    if (lane < 16) {
      atomicAdd(&msum[nt * 16 + lane], s);
      atomicAdd(&msumsq[nt * 16 + lane], q);
    }
  }
}

// Pass 2: y -> BN -> relu -> @W2 + b2 -> log_softmax, write output
__global__ void m2_pass2_kernel(const float* __restrict__ x2n,
                                const _Float16* __restrict__ w1f, const float* __restrict__ b1,
                                const float* __restrict__ scale, const float* __restrict__ shift,
                                const float* __restrict__ w2, const float* __restrict__ b2,
                                float* __restrict__ out) {
  __shared__ float H2[16 * 128];
  const int lane = threadIdx.x, hi = lane >> 4, lm = lane & 15;
  const int r0 = blockIdx.x * 16;
  const float *xi, *xz;
  m2_row_ptrs(x2n, r0 + lm, xi, xz);
  v16h a0 = m2_a_frag(xi, xz, 0, hi);
  v16h a1 = m2_a_frag(xi, xz, 1, hi);
#pragma unroll
  for (int nt = 0; nt < 8; ++nt) {
    v8f acc = {};
    acc = WMMA_F16(a0, load_wf(w1f, nt, 0, 2, lane), acc);
    acc = WMMA_F16(a1, load_wf(w1f, nt, 1, 2, lane), acc);
    int col = nt * 16 + lm;
    float bias = b1[col], sc = scale[col], sh = shift[col];
#pragma unroll
    for (int rr = 0; rr < 8; ++rr) {
      int m = rr + 8 * hi;
      H2[m * 128 + col] = fmaxf((acc[rr] + bias) * sc + sh, 0.0f);
    }
  }
  __syncthreads();
  if (lane < 16) {
    int r = r0 + lane;
    float o0 = b2[0], o1 = b2[1];
    for (int c = 0; c < HM; ++c) {
      float v = H2[lane * 128 + c];
      o0 += v * w2[c * 2 + 0];
      o1 += v * w2[c * 2 + 1];
    }
    float mx = fmaxf(o0, o1);
    float l  = mx + logf(expf(o0 - mx) + expf(o1 - mx));
    out[(size_t)r * 2 + 0] = o0 - l;
    out[(size_t)r * 2 + 1] = o1 - l;
  }
}

// ---------------------------------------------------------------------------
// Host launcher
// ---------------------------------------------------------------------------
extern "C" void kernel_launch(void* const* d_in, const int* in_sizes, int n_in,
                              void* d_out, int out_size, void* d_ws, size_t ws_size,
                              hipStream_t stream) {
  (void)in_sizes; (void)n_in; (void)out_size; (void)ws_size;
  const float* x        = (const float*)d_in[0];
  const int*   ei       = (const int*)  d_in[1];
  const float* ea       = (const float*)d_in[2];
  // d_in[3] = batch (implicit via contiguous layout)
  const float* ef_w1    = (const float*)d_in[4];
  const float* ef_b1    = (const float*)d_in[5];
  const float* ef_w2    = (const float*)d_in[6];
  const float* ef_b2    = (const float*)d_in[7];
  const float* ef_wroot = (const float*)d_in[8];
  const float* bn1_g    = (const float*)d_in[9];
  const float* bn1_b    = (const float*)d_in[10];
  const float* ec_w1    = (const float*)d_in[11];
  const float* ec_b1    = (const float*)d_in[12];
  const float* ec_w2    = (const float*)d_in[13];
  const float* ec_b2    = (const float*)d_in[14];
  const float* bn2_g    = (const float*)d_in[15];
  const float* bn2_b    = (const float*)d_in[16];
  const float* m2_w1    = (const float*)d_in[17];
  const float* m2_b1    = (const float*)d_in[18];
  const float* m2_bng   = (const float*)d_in[19];
  const float* m2_bnb   = (const float*)d_in[20];
  const float* m2_w2    = (const float*)d_in[21];
  const float* m2_b2    = (const float*)d_in[22];
  float* out = (float*)d_out;

  // --- workspace layout (256B aligned slices) ---
  char* ws = (char*)d_ws;
  size_t off = 0;
  auto take = [&](size_t bytes) -> char* {
    char* p = ws + off;
    off = (off + bytes + 255) & ~(size_t)255;
    return p;
  };
  float*     x1acc  = (float*)    take(sizeof(float) * NNODE * HDIM);
  float*     x1raw  = (float*)    take(sizeof(float) * NNODE * HDIM);
  float*     x1n    = (float*)    take(sizeof(float) * NNODE * HDIM);
  _Float16*  x1f    = (_Float16*) take(sizeof(_Float16) * NNODE * HDIM);  // fragment-major
  float*     sqn    = (float*)    take(sizeof(float) * NNODE);
  int*       nbr    = (int*)      take(sizeof(int) * NNODE * KNN);
  unsigned*  x2bits = (unsigned*) take(sizeof(unsigned) * NNODE * HDIM);
  float*     x2     = (float*)    take(sizeof(float) * NNODE * HDIM);
  float*     x2n    = (float*)    take(sizeof(float) * NNODE * HDIM);
  _Float16*  w1pf   = (_Float16*) take(sizeof(_Float16) * 96 * 64);
  _Float16*  efw2f  = (_Float16*) take(sizeof(_Float16) * 64 * 64);
  _Float16*  wrootf = (_Float16*) take(sizeof(_Float16) * 32 * 64);
  _Float16*  ecw1f  = (_Float16*) take(sizeof(_Float16) * 128 * 64);
  _Float16*  ecw2f  = (_Float16*) take(sizeof(_Float16) * 64 * 64);
  _Float16*  m2w1f  = (_Float16*) take(sizeof(_Float16) * 64 * 128);
  float*     s1     = (float*)    take(sizeof(float) * HDIM);
  float*     q1     = (float*)    take(sizeof(float) * HDIM);
  float*     s2     = (float*)    take(sizeof(float) * HDIM);
  float*     q2     = (float*)    take(sizeof(float) * HDIM);
  float*     sm     = (float*)    take(sizeof(float) * HM);
  float*     qm     = (float*)    take(sizeof(float) * HM);
  float*     sc1    = (float*)    take(sizeof(float) * HDIM);
  float*     sh1    = (float*)    take(sizeof(float) * HDIM);
  float*     sc2    = (float*)    take(sizeof(float) * HDIM);
  float*     sh2    = (float*)    take(sizeof(float) * HDIM);
  float*     scm    = (float*)    take(sizeof(float) * HM);
  float*     shm    = (float*)    take(sizeof(float) * HM);

  const int elemBlocks = (NNODE * HDIM + 255) / 256;

  init_kernel<<<elemBlocks, 256, 0, stream>>>(x1acc, x2bits, sm, qm);
  // one-shot weight swizzles (fragment-major f16)
  swizzle_w_kernel<<<(96 * 64 + 255) / 256, 256, 0, stream>>>(ef_w1, w1pf, 96, 72, 64);
  swizzle_w_kernel<<<(64 * 64 + 255) / 256, 256, 0, stream>>>(ef_w2, efw2f, 64, 64, 64);
  swizzle_w_kernel<<<(32 * 64 + 255) / 256, 256, 0, stream>>>(ef_wroot, wrootf, 32, 32, 64);
  swizzle_w_kernel<<<(128 * 64 + 255) / 256, 256, 0, stream>>>(ec_w1, ecw1f, 128, 128, 64);
  swizzle_w_kernel<<<(64 * 64 + 255) / 256, 256, 0, stream>>>(ec_w2, ecw2f, 64, 64, 64);
  swizzle_w_kernel<<<(64 * 128 + 255) / 256, 256, 0, stream>>>(m2_w1, m2w1f, 64, 64, 128);

  // Layer 1
  edgefeat_kernel<<<NEDGE / 16, 32, 0, stream>>>(x, ei, ea, w1pf, ef_b1, efw2f, ef_b2, x1acc);
  root_kernel<<<NNODE / 16, 32, 0, stream>>>(x, wrootf, x1acc, x1raw);
  colstats_kernel<<<HDIM, 256, 0, stream>>>(x1raw, NNODE, HDIM, s1, q1);
  bn_finalize_kernel<<<1, HDIM, 0, stream>>>(s1, q1, bn1_g, bn1_b, 1.0f / NNODE, HDIM, sc1, sh1);
  bn_apply_x1_kernel<<<(NNODE + 255) / 256, 256, 0, stream>>>(x1raw, sc1, sh1, x1n, sqn);
  swizzle_x1_kernel<<<((NNODE / 16) * 64 + 255) / 256, 256, 0, stream>>>(x1n, x1f);

  // Dynamic KNN + EdgeConv
  knn_kernel<<<NNODE / 16, 32, 0, stream>>>(x1f, sqn, nbr);
  edgeconv_kernel<<<(2 * NNODE * KNN) / 16, 32, 0, stream>>>(x1n, nbr, ecw1f, ec_b1, ecw2f, ec_b2, x2bits);
  x2_finalize_kernel<<<elemBlocks, 256, 0, stream>>>(x2bits, x2);
  colstats_kernel<<<HDIM, 256, 0, stream>>>(x2, NNODE, HDIM, s2, q2);
  bn_finalize_kernel<<<1, HDIM, 0, stream>>>(s2, q2, bn2_g, bn2_b, 1.0f / NNODE, HDIM, sc2, sh2);
  bn_apply_x2_kernel<<<elemBlocks, 256, 0, stream>>>(x2, sc2, sh2, x2n);

  // model2 head (fused |g_i - g_z| rows, two passes for BatchNorm stats)
  m2_pass1_kernel<<<2048, 32, 0, stream>>>(x2n, m2w1f, m2_b1, sm, qm);
  bn_finalize_kernel<<<1, HM, 0, stream>>>(sm, qm, m2_bng, m2_bnb, 1.0f / (float)R2, HM, scm, shm);
  m2_pass2_kernel<<<TILES2, 32, 0, stream>>>(x2n, m2w1f, m2_b1, scm, shm, m2_w2, m2_b2, out);
}
